// LatentDecoder_91242285236859
// MI455X (gfx1250) — matrix operand order repaired and checked by reading, not compiled, not run on hardware
//
#include <hip/hip_runtime.h>
#include <math.h>

// ---------------------------------------------------------------------------
// LatentDecoder for MI455X (gfx1250, wave32, WMMA, async-to-LDS).
// Pipeline:
//   prep:  convert+transpose weights to bf16 [N][K]; fuse b_ih+b_hh
//   K1:    build xcat bf16 [B*T, 2D]  (embed gather, shift, concat c^T)
//   G1:    pre1 = relu(xcat @ W1 + b1)     (WMMA bf16/f32, async LDS staging)
//   G2:    pre  = pre1 @ W2 + b2
//   G3:    xg   = pre @ W_ih + (b_ih+b_hh) stored permuted as [T,B,4D] bf16
//   LSTM:  persistent kernel, 16 WGs x 4 waves = 64 waves, 1 grid barrier/step,
//          W_hh fragments register-resident (1024-VGPR file), cell state in
//          VGPRs for all 1000 steps, h double-buffered in L2 (global_load)
//   G5:    logits = hs @ W_out + b_out  stored permuted [B,K,T] fp32 -> d_out
// ---------------------------------------------------------------------------

typedef unsigned short u16;
typedef __attribute__((ext_vector_type(16))) __bf16    v16bf;
typedef __attribute__((ext_vector_type(8)))  float     v8f;
typedef __attribute__((ext_vector_type(4)))  unsigned  u32x4;   // POD 128-bit

#define AS1 __attribute__((address_space(1)))

union FragBF { u32x4 q[2]; v16bf v; };

__device__ __forceinline__ u16 f2bf(float f) {
  unsigned u = __builtin_bit_cast(unsigned, f);
  u += 0x7fffu + ((u >> 16) & 1u);           // round-to-nearest-even
  return (u16)(u >> 16);
}
__device__ __forceinline__ float bf2f(u16 s) {
  unsigned u = ((unsigned)s) << 16;
  return __builtin_bit_cast(float, u);
}
__device__ __forceinline__ v8f vzero() {
  v8f z = {0.f, 0.f, 0.f, 0.f, 0.f, 0.f, 0.f, 0.f};
  return z;
}
__device__ __forceinline__ float sigm(float x) { return 1.f / (1.f + __expf(-x)); }

// generic -> explicit global address space (value-preserving int cast)
template <typename T>
__device__ __forceinline__ const AS1 T* gptr(const void* p) {
  return (const AS1 T*)(unsigned long long)(size_t)p;
}
template <typename T>
__device__ __forceinline__ AS1 T* gptr_mut(void* p) {
  return (AS1 T*)(unsigned long long)(size_t)p;
}

// CDNA5 async copy: global memory -> LDS, 16 bytes per lane, ASYNCcnt-tracked.
__device__ __forceinline__ void async_b128(const void* gsrc, void* lds_dst) {
  unsigned ldsoff = (unsigned)(size_t)lds_dst;          // aperture low bits
  unsigned long long ga = (unsigned long long)(size_t)gsrc;
  asm volatile("global_load_async_to_lds_b128 %0, %1, off"
               :: "v"(ldsoff), "v"(ga)
               : "memory");
}
__device__ __forceinline__ void async_wait0() {
  asm volatile("s_wait_asynccnt 0x0" ::: "memory");
}

static constexpr int CB = 32;      // batch
static constexpr int CT = 1000;    // time
static constexpr int CD = 512;     // hidden
static constexpr int CK = 512;     // vocab

// ---- prep kernels ----------------------------------------------------------

// W fp32 [Kd][N] row-major  ->  Wt bf16 [N][Kd] (transposed)
__global__ void k_wcvt(const float* __restrict__ W, u16* __restrict__ Wt, int Kd, int N) {
  int j = blockIdx.x * blockDim.x + threadIdx.x;
  if (j >= Kd * N) return;
  int n = j / Kd, k = j - n * Kd;
  Wt[j] = f2bf(W[(size_t)k * N + n]);
}

__global__ void k_bias4(const float* __restrict__ a, const float* __restrict__ b,
                        float* __restrict__ o, int n) {
  int i = blockIdx.x * blockDim.x + threadIdx.x;
  if (i < n) o[i] = a[i] + b[i];
}

// xcat[b*T+t, 0:D]   = emb[zi[b,t-1]]   (zeros at t==0)
// xcat[b*T+t, D:2D]  = c[b, :, t]
__global__ void k_buildx(const int* __restrict__ zi, const float* __restrict__ c,
                         const float* __restrict__ emb, u16* __restrict__ xcat) {
  int id = blockIdx.x * blockDim.x + threadIdx.x;
  const int total = CB * CT * (2 * CD);
  if (id >= total) return;
  int d2 = id & 1023;                 // 2D == 1024
  int m  = id >> 10;
  int b  = m / CT, t = m - b * CT;
  float v;
  if (d2 < CD) {
    v = (t == 0) ? 0.f : emb[(size_t)zi[b * CT + t - 1] * CD + d2];
  } else {
    v = c[((size_t)b * CD + (d2 - CD)) * CT + t];
  }
  xcat[id] = f2bf(v);
}

__global__ void k_init(u16* __restrict__ h0, int* __restrict__ counter) {
  int i = blockIdx.x * blockDim.x + threadIdx.x;
  if (i == 0) *counter = 0;
  if (i < CB * CD) h0[i] = 0;
}

// ---- generic WMMA GEMM:  C = A[M,Kd](bf16) x Bt[N,Kd](bf16)^T + bias -------
// MODE 0: relu, bf16 out row-major        MODE 1: bf16 out row-major
// MODE 2: bf16 out permuted (b*T+t -> t*B+b) for xg
// MODE 3: fp32 out permuted (t*B+b row -> [b, n, t]) for logits
template <int MODE>
__global__ __launch_bounds__(256) void k_gemm(const u16* __restrict__ A_,
                                              const u16* __restrict__ Bt_,
                                              const float* __restrict__ bias,
                                              void* __restrict__ Out,
                                              int M, int Kd, int N) {
  __shared__ u16 As[128 * 32];   // [m][k]
  __shared__ u16 Bs[64 * 32];    // [n][k]  (pre-transposed weights)
  const AS1 u16* A  = gptr<u16>(A_);
  const AS1 u16* Bt = gptr<u16>(Bt_);
  const int tileN = blockIdx.x * 64;
  const int tileM = blockIdx.y * 128;
  const int tid  = threadIdx.x;
  const int wave = tid >> 5, lane = tid & 31;
  const int wm = wave & 3, wn = wave >> 2;          // 4 x 2 waves -> 128 x 64
  const int col_l = lane & 15;
  const int rbase = (lane >> 4) * 8;

  // per-thread staging coordinates
  const int sa_r = tid >> 1, sa_h = (tid & 1) * 16;  // A: 16 bf16 (2x b128)
  const int sb_n = tid >> 2, sb_k = (tid & 3) * 8;   // B: 8 bf16  (1x b128)

  v8f acc[2][2] = { {vzero(), vzero()}, {vzero(), vzero()} };

  for (int k0 = 0; k0 < Kd; k0 += 32) {
    // async stage: global -> LDS directly (no VGPR round trip), ASYNCcnt
    {
      const AS1 u16* ga = A + (size_t)(tileM + sa_r) * Kd + k0 + sa_h;
      u16* la = As + sa_r * 32 + sa_h;
      async_b128((const void*)(size_t)(unsigned long long)ga, la);
      async_b128((const void*)(size_t)((unsigned long long)ga + 16), la + 8);
      const AS1 u16* gb = Bt + (size_t)(tileN + sb_n) * Kd + k0 + sb_k;
      async_b128((const void*)(size_t)(unsigned long long)gb, Bs + sb_n * 32 + sb_k);
    }
    async_wait0();
    __syncthreads();

    FragBF a[2], b[2];
#pragma unroll
    for (int mi = 0; mi < 2; ++mi) {
      const u16* p = As + (wm * 32 + mi * 16 + col_l) * 32 + rbase;
      a[mi].q[0] = *(const u32x4*)p;           // K = kb .. kb+7
      a[mi].q[1] = *(const u32x4*)(p + 16);    // K = kb+16 .. kb+23
    }
#pragma unroll
    for (int ni = 0; ni < 2; ++ni) {
      const u16* p = Bs + (wn * 32 + ni * 16 + col_l) * 32 + rbase;
      b[ni].q[0] = *(const u32x4*)p;           // K = kb .. kb+7
      b[ni].q[1] = *(const u32x4*)(p + 16);    // K = kb+16 .. kb+23
    }
#pragma unroll
    for (int mi = 0; mi < 2; ++mi)
#pragma unroll
      for (int ni = 0; ni < 2; ++ni)
        acc[mi][ni] = __builtin_amdgcn_wmma_f32_16x16x32_bf16(
            false, a[mi].v, false, b[ni].v, (short)0, acc[mi][ni], false, false);
    __syncthreads();
  }

  // fused epilogue
#pragma unroll
  for (int mi = 0; mi < 2; ++mi) {
#pragma unroll
    for (int ni = 0; ni < 2; ++ni) {
      const int gcol = tileN + wn * 32 + ni * 16 + col_l;
      const float bv = bias ? bias[gcol] : 0.f;
#pragma unroll
      for (int r = 0; r < 8; ++r) {
        const int grow = tileM + wm * 32 + mi * 16 + rbase + r;
        float v = acc[mi][ni][r] + bv;
        if (MODE == 0) v = v > 0.f ? v : 0.f;
        if (MODE == 0 || MODE == 1) {
          gptr_mut<u16>(Out)[(size_t)grow * N + gcol] = f2bf(v);
        } else if (MODE == 2) {               // row = b*T+t  ->  [t, b, n]
          int bb = grow / CT, tt = grow - bb * CT;
          gptr_mut<u16>(Out)[((size_t)tt * CB + bb) * N + gcol] = f2bf(v);
        } else {                              // row = t*B+b  ->  [b, n, t]
          int tt = grow / CB, bb = grow - tt * CB;
          gptr_mut<float>(Out)[((size_t)bb * N + gcol) * CT + tt] = v;
        }
      }
    }
  }
}

// ---- persistent LSTM recurrence -------------------------------------------
// 16 workgroups x 128 threads (4 waves) = 64 waves; wave w: m = w>>5 (batch
// half), nt = w&31 (hidden 16-col tile). Each wave accumulates all 4 gate
// tiles for its (m,nt) so the cell update is wave-local; W_hh fragments are
// loop-invariant -> compiler keeps them register-resident via the 1024-VGPR
// file; cell state stays in VGPRs for the whole scan. One device barrier/step.
__global__ __launch_bounds__(128) void k_lstm(const u16* __restrict__ xg_,
                                              const u16* __restrict__ Whht_,
                                              u16* __restrict__ h0_,
                                              u16* __restrict__ h1_,
                                              u16* __restrict__ hs_,
                                              int* __restrict__ counter) {
  const AS1 u16* xg   = gptr<u16>(xg_);
  const AS1 u16* Whht = gptr<u16>(Whht_);
  AS1 u16* hb0 = gptr_mut<u16>(h0_);
  AS1 u16* hb1 = gptr_mut<u16>(h1_);
  AS1 u16* hs  = gptr_mut<u16>(hs_);

  const int wave = blockIdx.x * 4 + (threadIdx.x >> 5);   // 0..63
  const int lane = threadIdx.x & 31;
  const int m = wave >> 5, nt = wave & 31;
  const int col_l = lane & 15;
  const int rbase = (lane >> 4) * 8;
  const int arow = m * 16 + col_l;                        // A-fragment row

  v8f cst = vzero();                                      // cell state (VGPRs)

  for (int t = 0; t < CT; ++t) {
    const AS1 u16* hin = (t & 1) ? hb1 : hb0;
    AS1 u16* hout      = (t & 1) ? hb0 : hb1;

    v8f acc[4] = { vzero(), vzero(), vzero(), vzero() };
    for (int kt = 0; kt < 16; ++kt) {                     // K = 512 / 32
      FragBF a;
      const AS1 u16* pa = hin + arow * CD + kt * 32 + rbase;
      a.q[0] = *(const AS1 u32x4*)pa;
      a.q[1] = *(const AS1 u32x4*)(pa + 16);
#pragma unroll
      for (int g = 0; g < 4; ++g) {
        FragBF b;
        const AS1 u16* pb = Whht + (size_t)(g * CD + nt * 16 + col_l) * CD
                                 + kt * 32 + rbase;
        b.q[0] = *(const AS1 u32x4*)pb;
        b.q[1] = *(const AS1 u32x4*)(pb + 16);
        acc[g] = __builtin_amdgcn_wmma_f32_16x16x32_bf16(
            false, a.v, false, b.v, (short)0, acc[g], false, false);
      }
    }

#pragma unroll
    for (int r = 0; r < 8; ++r) {
      const int brow = m * 16 + rbase + r;
      const int ch = nt * 16 + col_l;
      const AS1 u16* xr = xg + ((size_t)t * CB + brow) * (4 * CD);
      float gi = sigm (acc[0][r] + bf2f(xr[ch]));
      float gf = sigm (acc[1][r] + bf2f(xr[CD + ch]));
      float gg = tanhf(acc[2][r] + bf2f(xr[2 * CD + ch]));
      float go = sigm (acc[3][r] + bf2f(xr[3 * CD + ch]));
      float cc = gf * cst[r] + gi * gg;
      cst[r] = cc;
      u16 hb = f2bf(go * tanhf(cc));
      hout[brow * CD + ch] = hb;
      hs[((size_t)t * CB + brow) * CD + ch] = hb;
    }

    // device-wide barrier (16 resident WGs)
    __threadfence();
    __syncthreads();
    if (threadIdx.x == 0) {
      __hip_atomic_fetch_add(counter, 1, __ATOMIC_RELEASE, __HIP_MEMORY_SCOPE_AGENT);
      while (__hip_atomic_load(counter, __ATOMIC_ACQUIRE, __HIP_MEMORY_SCOPE_AGENT)
             < 16 * (t + 1))
        __builtin_amdgcn_s_sleep(1);
    }
    __syncthreads();
  }
}

// ---------------------------------------------------------------------------

extern "C" void kernel_launch(void* const* d_in, const int* in_sizes, int n_in,
                              void* d_out, int out_size, void* d_ws, size_t ws_size,
                              hipStream_t stream) {
  const int*   zi    = (const int*)  d_in[0];
  const float* c     = (const float*)d_in[1];
  const float* emb   = (const float*)d_in[2];
  const float* W1    = (const float*)d_in[3];
  const float* b1    = (const float*)d_in[4];
  const float* W2    = (const float*)d_in[5];
  const float* b2    = (const float*)d_in[6];
  const float* W_ih  = (const float*)d_in[7];
  const float* b_ih  = (const float*)d_in[8];
  const float* W_hh  = (const float*)d_in[9];
  const float* b_hh  = (const float*)d_in[10];
  const float* W_out = (const float*)d_in[11];
  const float* b_out = (const float*)d_in[12];

  char* ws = (char*)d_ws;
  size_t off = 0;
  auto alloc = [&](size_t bytes) -> void* {
    off = (off + 255) & ~(size_t)255;
    void* p = ws + off;
    off += bytes;
    return p;
  };

  const int M = CB * CT;                       // 32000
  u16*   Wt1   = (u16*)  alloc((size_t)(2 * CD) * CD * 2);   // [512][1024]
  u16*   Wt2   = (u16*)  alloc((size_t)CD * CD * 2);         // [512][512]
  u16*   Wtih  = (u16*)  alloc((size_t)(4 * CD) * CD * 2);   // [2048][512]
  u16*   Wthh  = (u16*)  alloc((size_t)(4 * CD) * CD * 2);   // [2048][512]
  u16*   Wtout = (u16*)  alloc((size_t)CK * CD * 2);         // [512][512]
  float* bias4 = (float*)alloc((size_t)(4 * CD) * 4);
  u16*   xcat  = (u16*)  alloc((size_t)M * (2 * CD) * 2);    // 65.5 MB
  u16*   pre1  = (u16*)  alloc((size_t)M * CD * 2);          // 32.8 MB
  u16*   pre2  = (u16*)  alloc((size_t)M * CD * 2);          // 32.8 MB
  u16*   xg    = (u16*)  alloc((size_t)M * (4 * CD) * 2);    // 131 MB [T,B,4D]
  u16*   hs    = (u16*)  alloc((size_t)M * CD * 2);          // 32.8 MB [T,B,D]
  u16*   h0b   = (u16*)  alloc((size_t)CB * CD * 2);
  u16*   h1b   = (u16*)  alloc((size_t)CB * CD * 2);
  int*   cnt   = (int*)  alloc(256);
  (void)ws_size; (void)in_sizes; (void)n_in; (void)out_size;

  const int TPB = 256;
  // weight convert + transpose (bf16 [N][K])
  k_wcvt<<<((2 * CD) * CD + TPB - 1) / TPB, TPB, 0, stream>>>(W1, Wt1, 2 * CD, CD);
  k_wcvt<<<(CD * CD + TPB - 1) / TPB, TPB, 0, stream>>>(W2, Wt2, CD, CD);
  k_wcvt<<<(CD * (4 * CD) + TPB - 1) / TPB, TPB, 0, stream>>>(W_ih, Wtih, CD, 4 * CD);
  k_wcvt<<<(CD * (4 * CD) + TPB - 1) / TPB, TPB, 0, stream>>>(W_hh, Wthh, CD, 4 * CD);
  k_wcvt<<<(CD * CK + TPB - 1) / TPB, TPB, 0, stream>>>(W_out, Wtout, CD, CK);
  k_bias4<<<(4 * CD + TPB - 1) / TPB, TPB, 0, stream>>>(b_ih, b_hh, bias4, 4 * CD);
  // inputs: gather/shift/concat
  k_buildx<<<((size_t)M * (2 * CD) + TPB - 1) / TPB, TPB, 0, stream>>>(zi, c, emb, xcat);
  k_init<<<(CB * CD + TPB - 1) / TPB, TPB, 0, stream>>>(h0b, cnt);

  // batched GEMMs (WMMA bf16 / f32 accum, async-to-LDS staging)
  dim3 blk(256);
  dim3 g1(CD / 64, M / 128);
  k_gemm<0><<<g1, blk, 0, stream>>>(xcat, Wt1, b1, pre1, M, 2 * CD, CD);
  dim3 g2(CD / 64, M / 128);
  k_gemm<1><<<g2, blk, 0, stream>>>(pre1, Wt2, b2, pre2, M, CD, CD);
  dim3 g3((4 * CD) / 64, M / 128);
  k_gemm<2><<<g3, blk, 0, stream>>>(pre2, Wtih, bias4, xg, M, CD, 4 * CD);

  // sequential recurrence: persistent kernel, 1 barrier per step
  k_lstm<<<16, 128, 0, stream>>>(xg, Wthh, h0b, h1b, hs, cnt);

  // output projection with fused [B,K,T] permute, fp32 out
  dim3 g5(CK / 64, M / 128);
  k_gemm<3><<<g5, blk, 0, stream>>>(hs, Wtout, b_out, (float*)d_out, M, CD, CK);
}
